// MultiHeadAttentionLayer_15075335209548
// MI455X (gfx1250) — matrix operand order repaired and checked by reading, not compiled, not run on hardware
//
#include <hip/hip_runtime.h>

// ---------------------------------------------------------------------------
// MI455X (gfx1250, wave32) fused multi-head attention, bf16 WMMA path.
//
// Pipeline (all intermediates in d_ws):
//   1) convert query + 4 weight matrices f32 -> bf16
//   2) Q = (X Wq^T + bq)/8   -> bf16 [B,H,S,D]      (scale folded into Q)
//      K = (X Wk^T + bk)     -> bf16 [B,H,S,D]
//      V = (X Wv^T + bv)     -> bf16 [B,H,D,S]      (transposed for PV GEMM)
//   3) attn_stats: per (b,h,j) column max & 1/sum of exp over the QUERY axis
//      (softmax axis=2!), mask applied multiplicatively to energy.
//   4) attn_pv: recompute e^T tiles, normalize, x^T += V^T P^T  (flash-style,
//      never materializes the 1 GiB energy tensor)
//   5) out = Xattn Wo^T + bo  (f32 output)
//
// All matrix math uses v_wmma_f32_16x16x32_bf16 (f32 accumulate).
// ---------------------------------------------------------------------------

typedef __attribute__((ext_vector_type(16))) __bf16       v16bf;
typedef __attribute__((ext_vector_type(8)))  float        v8f;
typedef __attribute__((ext_vector_type(4)))  unsigned int v4u;

#define BATCH 4
#define SEQ   2048
#define HEADS 16
#define HDIM  64
#define HID   1024
#define MTOT  (BATCH * SEQ)          // 8192

// workspace layout (bytes)
#define OFF_XBF   ((size_t)0)                       // 8192*1024 bf16 = 16 MB
#define OFF_WQ    (OFF_XBF + (size_t)MTOT * HID * 2)
#define OFF_WK    (OFF_WQ  + (size_t)HID * HID * 2)
#define OFF_WV    (OFF_WK  + (size_t)HID * HID * 2)
#define OFF_WO    (OFF_WV  + (size_t)HID * HID * 2)
#define OFF_QBF   (OFF_WO  + (size_t)HID * HID * 2) // [B,H,S,D] bf16
#define OFF_KBF   (OFF_QBF + (size_t)MTOT * HID * 2)
#define OFF_VT    (OFF_KBF + (size_t)MTOT * HID * 2) // [B,H,D,S] bf16
#define OFF_XAT   (OFF_VT  + (size_t)MTOT * HID * 2) // [M,HID] bf16
#define OFF_CMAX  (OFF_XAT + (size_t)MTOT * HID * 2) // [B*H*S] f32
#define OFF_CREC  (OFF_CMAX + (size_t)BATCH * HEADS * SEQ * 4)

union Frag {
    v16bf v;
    v4u   q[2];
    unsigned int u[8];
};

__device__ __forceinline__ unsigned short f2bf(float f) {
    unsigned int u = __float_as_uint(f);
    u += 0x7FFFu + ((u >> 16) & 1u);      // round-to-nearest-even
    return (unsigned short)(u >> 16);
}
__device__ __forceinline__ unsigned int pack2(float a, float b) {
    return (unsigned int)f2bf(a) | ((unsigned int)f2bf(b) << 16);
}

// A-matrix fragment, 16x32 bf16 (MxK), row-major src with row stride ld.
// Contiguous k order used here (same as loadB): lanes 0-15: M=lane, K = k0+{0..15};
//             lanes 16-31: M=lane-16, K = k0+{16..31}.
__device__ __forceinline__ Frag loadA(const unsigned short* base, int row,
                                      int ld, int k0, int lane) {
    int lm = lane & 15, hf = lane >> 4;
    const unsigned short* p = base + (size_t)(row + lm) * ld;
    Frag f;
    f.q[0] = *(const v4u*)(p + k0 + hf * 16);
    f.q[1] = *(const v4u*)(p + k0 + 8 + hf * 16);
    return f;
}

// B-matrix fragment, 32x16 bf16 (KxN), where column n of B is row n of src
// (i.e. B[k,n] = src[n, k0+k]).  Contiguous k order used here: lanes 0-15 hold K=k0..k0+15
// (VGPR v: K=2v,2v+1), lanes 16-31 hold K=k0+16..k0+31.
__device__ __forceinline__ Frag loadB(const unsigned short* base, int nrow,
                                      int ld, int k0, int lane) {
    int lm = lane & 15, hf = lane >> 4;
    const unsigned short* p = base + (size_t)(nrow + lm) * ld + k0 + hf * 16;
    Frag f;
    f.q[0] = *(const v4u*)(p);
    f.q[1] = *(const v4u*)(p + 8);
    return f;
}

__device__ __forceinline__ v8f wmma_bf16(const Frag& a, const Frag& b, v8f c) {
    return __builtin_amdgcn_wmma_f32_16x16x32_bf16(
        false, a.v, false, b.v, (short)0, c, false, false);
}

// ---------------------------------------------------------------------------
__global__ void f32_to_bf16_kernel(const float* __restrict__ src,
                                   unsigned short* __restrict__ dst, int n) {
    for (int i = blockIdx.x * blockDim.x + threadIdx.x; i < n;
         i += gridDim.x * blockDim.x)
        dst[i] = f2bf(src[i]);
}

// ---------------------------------------------------------------------------
// C[m,n] = sum_k X[m,k] * W[n,k] + bias[n], then scaled and stored per mode.
// mode 0: bf16 [B,H,S,D]  (Q, scale=1/8)
// mode 1: bf16 [B,H,S,D]  (K)
// mode 2: bf16 [B,H,D,S]  (V transposed)
// mode 3: f32  [M,HID]    (final output)
// One wave per block; each wave computes a 16(M) x 64(N) tile.
__global__ __launch_bounds__(32)
void gemm_bias_kernel(const unsigned short* __restrict__ Xb,
                      const unsigned short* __restrict__ Wb,
                      const float* __restrict__ bias,
                      void* __restrict__ out, int mode, float scale) {
    const int lane = threadIdx.x, lm = lane & 15, hf = lane >> 4;
    const int m0 = blockIdx.x * 16;
    const int n0 = blockIdx.y * 64;

    v8f acc[4];
#pragma unroll
    for (int s = 0; s < 4; ++s) {
        float bv = bias[n0 + s * 16 + lm];
#pragma unroll
        for (int r = 0; r < 8; ++r) acc[s][r] = bv;
    }

    for (int k0 = 0; k0 < HID; k0 += 32) {
        Frag a = loadA(Xb, m0, HID, k0, lane);
#pragma unroll
        for (int s = 0; s < 4; ++s) {
            Frag b = loadB(Wb, n0 + s * 16, HID, k0, lane);
            acc[s] = wmma_bf16(a, b, acc[s]);
        }
    }

    const int bb = m0 / SEQ;          // batch (tile never crosses batch)
    const int s0 = m0 % SEQ;
    if (mode == 3) {
        float* O = (float*)out;
#pragma unroll
        for (int s = 0; s < 4; ++s) {
            int n = n0 + s * 16 + lm;
#pragma unroll
            for (int r = 0; r < 8; ++r) {
                int m = m0 + r + hf * 8;
                O[(size_t)m * HID + n] = acc[s][r] * scale;
            }
        }
    } else if (mode == 2) {           // V^T: [B,H,D,S], 8 consecutive s -> b128
        unsigned short* O = (unsigned short*)out;
#pragma unroll
        for (int s = 0; s < 4; ++s) {
            int n = n0 + s * 16 + lm;
            int h = n >> 6, d = n & 63;
            size_t base = ((size_t)(bb * HEADS + h) * HDIM + d) * SEQ + s0 + hf * 8;
            v4u st;
            st.x = pack2(acc[s][0] * scale, acc[s][1] * scale);
            st.y = pack2(acc[s][2] * scale, acc[s][3] * scale);
            st.z = pack2(acc[s][4] * scale, acc[s][5] * scale);
            st.w = pack2(acc[s][6] * scale, acc[s][7] * scale);
            *(v4u*)(O + base) = st;
        }
    } else {                          // Q/K: [B,H,S,D]
        unsigned short* O = (unsigned short*)out;
#pragma unroll
        for (int s = 0; s < 4; ++s) {
            int n = n0 + s * 16 + lm;
            int h = n >> 6, d = n & 63;
#pragma unroll
            for (int r = 0; r < 8; ++r) {
                int ss = s0 + r + hf * 8;
                O[((size_t)(bb * HEADS + h) * SEQ + ss) * HDIM + d] =
                    f2bf(acc[s][r] * scale);
            }
        }
    }
}

// ---------------------------------------------------------------------------
// Column softmax stats over the QUERY axis: for each (b,h,j):
//   cmax[j] = max_i e[i,j],  crec[j] = 1 / sum_i exp(e[i,j]-cmax[j])
// with e[i,j] = (q_i.k_j) * m_i * m_j  (scale folded into Q).
// e^T tiles (M=j, N=i) so the reduction over i is a lane reduction.
__global__ __launch_bounds__(32)
void attn_stats_kernel(const unsigned short* __restrict__ Qb,
                       const unsigned short* __restrict__ Kb,
                       const unsigned char* __restrict__ mask,
                       float* __restrict__ cmax, float* __restrict__ crec) {
    const int lane = threadIdx.x, lm = lane & 15, hf = lane >> 4;
    const int jT = blockIdx.x & 127;
    const int bh = blockIdx.x >> 7;
    const int bb = bh >> 4;
    const int j0 = jT * 16;
    const unsigned short* Qbh = Qb + (size_t)bh * SEQ * HDIM;
    const unsigned short* Kbh = Kb + (size_t)bh * SEQ * HDIM;

    Frag a0 = loadA(Kbh, j0, HDIM, 0, lane);      // invariant over i
    Frag a1 = loadA(Kbh, j0, HDIM, 32, lane);

    float mj[8];
#pragma unroll
    for (int r = 0; r < 8; ++r)
        mj[r] = mask[bb * SEQ + j0 + r + hf * 8] ? 1.0f : 0.0f;

    float rmax[8], rsum[8];
#pragma unroll
    for (int r = 0; r < 8; ++r) { rmax[r] = -3.0e38f; rsum[r] = 0.0f; }

    for (int i0 = 0; i0 < SEQ; i0 += 16) {
        Frag b0 = loadB(Qbh, i0, HDIM, 0, lane);
        Frag b1 = loadB(Qbh, i0, HDIM, 32, lane);
        v8f e = {};
        e = wmma_bf16(a0, b0, e);
        e = wmma_bf16(a1, b1, e);
        float mi = mask[bb * SEQ + i0 + lm] ? 1.0f : 0.0f;
#pragma unroll
        for (int r = 0; r < 8; ++r) {
            float ev = e[r] * mi * mj[r];
            float nm = fmaxf(rmax[r], ev);
            rsum[r] = rsum[r] * __expf(rmax[r] - nm) + __expf(ev - nm);
            rmax[r] = nm;
        }
    }
    // reduce over the 16 lanes of each half (i axis)
#pragma unroll
    for (int off = 8; off >= 1; off >>= 1) {
#pragma unroll
        for (int r = 0; r < 8; ++r) {
            float om = __shfl_xor(rmax[r], off, 32);
            float os = __shfl_xor(rsum[r], off, 32);
            float nm = fmaxf(rmax[r], om);
            rsum[r] = rsum[r] * __expf(rmax[r] - nm) + os * __expf(om - nm);
            rmax[r] = nm;
        }
    }
    float sm = rmax[0], ssum = rsum[0];
#pragma unroll
    for (int r = 1; r < 8; ++r)
        if ((lane & 7) == r) { sm = rmax[r]; ssum = rsum[r]; }
    if ((lane & 8) == 0) {
        int j = j0 + (lane & 7) + ((lane >> 4) << 3);
        cmax[(size_t)bh * SEQ + j] = sm;
        crec[(size_t)bh * SEQ + j] = 1.0f / ssum;
    }
}

// ---------------------------------------------------------------------------
// x^T[d,i] = sum_j V^T[d,j] * P^T[j,i],  P[i,j] = exp(e[i,j]-cmax[j])*crec[j]
// Streams over j, recomputing e^T tiles; transposes the f32 C fragment into a
// bf16 B fragment with one shfl_xor(16) half swap.
__global__ __launch_bounds__(32)
void attn_pv_kernel(const unsigned short* __restrict__ Qb,
                    const unsigned short* __restrict__ Kb,
                    const unsigned short* __restrict__ Vt,
                    const unsigned char* __restrict__ mask,
                    const float* __restrict__ cmax,
                    const float* __restrict__ crec,
                    unsigned short* __restrict__ Xat) {
    const int lane = threadIdx.x, lm = lane & 15, hf = lane >> 4;
    const int iT = blockIdx.x & 127;
    const int bh = blockIdx.x >> 7;
    const int bb = bh >> 4, h = bh & 15;
    const int i0 = iT * 16;
    const unsigned short* Qbh = Qb + (size_t)bh * SEQ * HDIM;
    const unsigned short* Kbh = Kb + (size_t)bh * SEQ * HDIM;
    const unsigned short* Vth = Vt + (size_t)bh * HDIM * SEQ;
    const float* cmx = cmax + (size_t)bh * SEQ;
    const float* crc = crec + (size_t)bh * SEQ;

    Frag q0 = loadB(Qbh, i0, HDIM, 0, lane);      // invariant over j
    Frag q1 = loadB(Qbh, i0, HDIM, 32, lane);
    float mi = mask[bb * SEQ + i0 + lm] ? 1.0f : 0.0f;

    v8f xa[4];
#pragma unroll
    for (int t = 0; t < 4; ++t) xa[t] = {};

    for (int j0 = 0; j0 < SEQ; j0 += 32) {
        float P1[8], P2[8];
#pragma unroll
        for (int t = 0; t < 2; ++t) {
            int jt = j0 + t * 16;
            Frag a0 = loadA(Kbh, jt, HDIM, 0, lane);
            Frag a1 = loadA(Kbh, jt, HDIM, 32, lane);
            v8f e = {};
            e = wmma_bf16(a0, q0, e);
            e = wmma_bf16(a1, q1, e);
            float* Pt = t ? P2 : P1;
#pragma unroll
            for (int r = 0; r < 8; ++r) {
                int j = jt + r + hf * 8;
                float mjv = mask[bb * SEQ + j] ? 1.0f : 0.0f;
                float ev = e[r] * mi * mjv;
                Pt[r] = __expf(ev - cmx[j]) * crc[j];
            }
        }
        // Build B fragment (K=j local 0..31, N=i): half-swap between the two
        // e^T tiles so each lane holds the 16 j-values its B slot needs.
        float pl[16];
#pragma unroll
        for (int r = 0; r < 8; ++r) {
            float y1 = __shfl_xor(P1[r], 16, 32);
            float y2 = __shfl_xor(P2[r], 16, 32);
            pl[r]     = hf ? y2 : P1[r];   // j_local r   (low) / 16+r (high)
            pl[8 + r] = hf ? P2[r] : y1;   // j_local 8+r (low) / 24+r (high)
        }
        Frag bp;
#pragma unroll
        for (int v = 0; v < 8; ++v) bp.u[v] = pack2(pl[2 * v], pl[2 * v + 1]);
#pragma unroll
        for (int dT = 0; dT < 4; ++dT) {
            Frag av = loadA(Vth, dT * 16, SEQ, j0, lane);
            xa[dT] = wmma_bf16(av, bp, xa[dT]);
        }
    }
    // store x as bf16 [M, HID]; 8 consecutive d per lane -> one b128 store
    size_t row = (size_t)(bb * SEQ + i0 + lm) * HID;
#pragma unroll
    for (int dT = 0; dT < 4; ++dT) {
        v4u st;
        st.x = pack2(xa[dT][0], xa[dT][1]);
        st.y = pack2(xa[dT][2], xa[dT][3]);
        st.z = pack2(xa[dT][4], xa[dT][5]);
        st.w = pack2(xa[dT][6], xa[dT][7]);
        *(v4u*)(Xat + row + h * HDIM + dT * 16 + hf * 8) = st;
    }
}

// ---------------------------------------------------------------------------
extern "C" void kernel_launch(void* const* d_in, const int* in_sizes, int n_in,
                              void* d_out, int out_size, void* d_ws, size_t ws_size,
                              hipStream_t stream) {
    const float*         query = (const float*)d_in[0];
    const unsigned char* mask  = (const unsigned char*)d_in[1]; // jnp.bool_: 1B
    const float* Wq = (const float*)d_in[2];
    const float* bq = (const float*)d_in[3];
    const float* Wk = (const float*)d_in[4];
    const float* bk = (const float*)d_in[5];
    const float* Wv = (const float*)d_in[6];
    const float* bv = (const float*)d_in[7];
    const float* Wo = (const float*)d_in[8];
    const float* bo = (const float*)d_in[9];

    char* w = (char*)d_ws;
    unsigned short* xbf = (unsigned short*)(w + OFF_XBF);
    unsigned short* wqb = (unsigned short*)(w + OFF_WQ);
    unsigned short* wkb = (unsigned short*)(w + OFF_WK);
    unsigned short* wvb = (unsigned short*)(w + OFF_WV);
    unsigned short* wob = (unsigned short*)(w + OFF_WO);
    unsigned short* qbf = (unsigned short*)(w + OFF_QBF);
    unsigned short* kbf = (unsigned short*)(w + OFF_KBF);
    unsigned short* vtb = (unsigned short*)(w + OFF_VT);
    unsigned short* xat = (unsigned short*)(w + OFF_XAT);
    float* cmx = (float*)(w + OFF_CMAX);
    float* crc = (float*)(w + OFF_CREC);

    // 1) f32 -> bf16 conversions
    f32_to_bf16_kernel<<<2048, 256, 0, stream>>>(query, xbf, MTOT * HID);
    f32_to_bf16_kernel<<<1024, 256, 0, stream>>>(Wq, wqb, HID * HID);
    f32_to_bf16_kernel<<<1024, 256, 0, stream>>>(Wk, wkb, HID * HID);
    f32_to_bf16_kernel<<<1024, 256, 0, stream>>>(Wv, wvb, HID * HID);
    f32_to_bf16_kernel<<<1024, 256, 0, stream>>>(Wo, wob, HID * HID);

    // 2) QKV projections (scale 1/sqrt(64) folded into Q)
    dim3 gg(MTOT / 16, HID / 64);
    gemm_bias_kernel<<<gg, 32, 0, stream>>>(xbf, wqb, bq, qbf, 0, 0.125f);
    gemm_bias_kernel<<<gg, 32, 0, stream>>>(xbf, wkb, bk, kbf, 1, 1.0f);
    gemm_bias_kernel<<<gg, 32, 0, stream>>>(xbf, wvb, bv, vtb, 2, 1.0f);

    // 3) column softmax stats (softmax over the query axis)
    int nblk = BATCH * HEADS * (SEQ / 16);   // 8192
    attn_stats_kernel<<<nblk, 32, 0, stream>>>(qbf, kbf, mask, cmx, crc);

    // 4) P @ V (recompute energy, flash style)
    attn_pv_kernel<<<nblk, 32, 0, stream>>>(qbf, kbf, vtb, mask, cmx, crc, xat);

    // 5) output projection (f32 out)
    gemm_bias_kernel<<<gg, 32, 0, stream>>>(xat, wob, bo, d_out, 3, 1.0f);
}
